// GNNE2C_88819923681894
// MI455X (gfx1250) — compile-verified, matrix operand order checked by reading, not run on hardware
//
#include <hip/hip_runtime.h>

typedef _Float16 f16;
typedef __attribute__((ext_vector_type(16))) _Float16 v16h;
typedef __attribute__((ext_vector_type(8)))  float    v8f;

#define DYN   96
#define STA   32
#define U9    9
#define NOBS  14
#define HZD   128
#define HID   200
#define BSZ   2048
#define TSTEPS 16
#define BT    32          // batch rows per workgroup
#define WAVES 8
#define THREADS 256

// padded GEMM dims
#define K1 160            // 129 -> 160, 5 k-slices
#define K2 224            // 200 -> 224, 7 k-slices

// packed-weight tiling (tiles are 32x16 f16 = 512 elements, B-fragment layout)
#define NK1 5
#define NT1 13            // 200 -> 208
#define NK2 7
#define NT2 13
#define NK3 7
#define NT3 8             // N = 128
#define NKA 387           // (128 rows + 1 bias row) * 96 / 32
#define NTA 6             // N = 96
#define NKB 37            // 9*128/32 + 1 bias slice
#define NTB 6
#define NKC 387
#define NTC 1             // N = 16 (14 padded)
#define NKD 37
#define NTD 1

#define SZ_W1 (NK1*NT1*512)
#define SZ_W2 (NK2*NT2*512)
#define SZ_W3 (NK3*NT3*512)
#define SZ_AT (NKA*NTA*512)
#define SZ_BT (NKB*NTB*512)
#define SZ_CT (NKC*NTC*512)
#define SZ_DT (NKD*NTD*512)

// ---------------- fragment helpers ----------------

// A/B 16-bit fragment k-map: element e on lane l -> k = (e&7) + 8*(l>>4) + 16*(e>>3)
__device__ __forceinline__ v16h load_a16(const f16* row, int lhi) {
    const f16* p = row + 8 * lhi;
    v16h r;
#pragma unroll
    for (int i = 0; i < 8; ++i) { r[i] = p[i]; r[8 + i] = p[16 + i]; }
    return r;
}

__device__ __forceinline__ v16h load_a32(const float* row, int lhi) {
    const float* p = row + 8 * lhi;
    v16h r;
#pragma unroll
    for (int i = 0; i < 8; ++i) { r[i] = (f16)p[i]; r[8 + i] = (f16)p[16 + i]; }
    return r;
}

__device__ __forceinline__ v16h scale16(v16h a, f16 s) {
    v16h r;
#pragma unroll
    for (int i = 0; i < 16; ++i) r[i] = a[i] * s;   // -> v_pk_mul_f16
    return r;
}

__device__ __forceinline__ v8f wmma32(v16h a, v16h b, v8f c) {
    return __builtin_amdgcn_wmma_f32_16x16x32_f16(false, a, false, b, (short)0, c,
                                                  false, false);
}

// ---------------- weight packing (f32 -> f16 B-fragment tiles) ----------------

__device__ __forceinline__ void pack_idx(long id, int nK, int& ks, int& nt, int& kl, int& nl) {
    int e = (int)(id & 15);
    int l = (int)((id >> 4) & 31);
    long tile = id >> 9;
    ks = (int)(tile % nK);
    nt = (int)(tile / nK);
    kl = (e & 7) + ((l >> 4) << 3) + ((e >> 3) << 4);
    nl = l & 15;
}

// plain row-major [Ko, No] weight
__global__ void pack_plain(const float* __restrict__ src, f16* __restrict__ dst,
                           int nK, int Ko, int No) {
    long id = (long)blockIdx.x * 256 + threadIdx.x;
    int ks, nt, kl, nl; pack_idx(id, nK, ks, nt, kl, nl);
    int k = ks * 32 + kl, n = nt * 16 + nl;
    float v = (k < Ko && n < No) ? src[(long)k * No + n] : 0.f;
    dst[id] = (f16)v;
}

// bilinear, j-fastest: Kidx = k*96 + j, N = i ; src[k][i*96+j]; k==Krows -> bias row
__global__ void pack_bilin_j(const float* __restrict__ src, const float* __restrict__ bias,
                             f16* __restrict__ dst, int nK, int Krows, int No) {
    long id = (long)blockIdx.x * 256 + threadIdx.x;
    int ks, nt, kl, nl; pack_idx(id, nK, ks, nt, kl, nl);
    int kidx = ks * 32 + kl, n = nt * 16 + nl;
    int k = kidx / 96, j = kidx % 96;
    float v = 0.f;
    if (n < No) v = (k < Krows) ? src[(long)k * (No * 96) + n * 96 + j] : bias[n * 96 + j];
    dst[id] = (f16)v;
}

// bilinear, k-fastest per u: Kidx = u*128 + k (Kidx<1152), then one bias slice (u = t < 9)
__global__ void pack_bilin_u(const float* __restrict__ src, const float* __restrict__ bias,
                             f16* __restrict__ dst, int nK, int No) {
    long id = (long)blockIdx.x * 256 + threadIdx.x;
    int ks, nt, kl, nl; pack_idx(id, nK, ks, nt, kl, nl);
    int kidx = ks * 32 + kl, n = nt * 16 + nl;
    float v = 0.f;
    if (n < No) {
        if (kidx < 1152) {
            int u = kidx >> 7, k = kidx & 127;
            v = src[(long)k * (No * 9) + n * 9 + u];
        } else {
            int t = kidx - 1152;
            if (t < 9) v = bias[n * 9 + t];
        }
    }
    dst[id] = (f16)v;
}

// ---------------- MLP GEMM (A from LDS f16, W packed, out to LDS f16) ----------------

__device__ __forceinline__ void mlp_gemm(const f16* A, int As, int nK,
                                         const f16* __restrict__ Wp, int nNt,
                                         const float* __restrict__ bias, int No,
                                         f16* Out, int Os, bool relu,
                                         int w, int lane) {
    const int lrow = lane & 15, lhi = lane >> 4;
    const int nTiles = 2 * nNt;                   // 2 M-tiles (BT = 32)
    for (int tdx = w; tdx < nTiles; tdx += WAVES) {
        int mt = tdx / nNt, nt = tdx % nNt;
        int n = nt * 16 + lrow;
        v8f acc = {};
        for (int ks = 0; ks < nK; ++ks) {
            v16h a = load_a16(&A[(mt * 16 + lrow) * As + ks * 32], lhi);
            const f16* bp = Wp + ((long)(nt * nK + ks)) * 512 + lane * 16;
            acc = wmma32(a, *(const v16h*)bp, acc);
        }
        float bb = (n < No) ? bias[n] : 0.f;
#pragma unroll
        for (int v = 0; v < 8; ++v) {
            int m = mt * 16 + v + 8 * lhi;        // C layout: lanes>=16 hold M+8
            float x = acc[v] + bb;
            if (relu) x = fmaxf(x, 0.f);
            Out[m * Os + n] = (f16)x;
        }
    }
}

// ---------------- main recurrent kernel ----------------

__launch_bounds__(THREADS)
__global__ void cond_lin_main(const float* __restrict__ zdyn, const float* __restrict__ zsta,
                              const float* __restrict__ dtg,  const float* __restrict__ Ug,
                              const float* __restrict__ b1,   const float* __restrict__ b2,
                              const float* __restrict__ b3,
                              const f16* __restrict__ W1p, const f16* __restrict__ W2p,
                              const f16* __restrict__ W3p, const f16* __restrict__ Atp,
                              const f16* __restrict__ Btp, const f16* __restrict__ Ctp,
                              const f16* __restrict__ Dtp,
                              float* __restrict__ Zout, float* __restrict__ Yout) {
    __shared__ f16   inp16[BT * K1];     // [z(96) | z_static(32) | dt(1) | 0-pad] per row
    __shared__ f16   h1[BT * K2];
    __shared__ f16   h2[BT * K2];
    __shared__ f16   hz16[BT * HZD];
    __shared__ float z2[BT * DYN];       // z_next accumulator (f32, atomically reduced)
    __shared__ f16   udt16[BT * 32];     // u*dt, padded to 32 cols
    __shared__ float yacc[BT * 16];

    const int tid = threadIdx.x;
    const int lane = tid & 31, w = tid >> 5;
    const int lrow = lane & 15, lhi = lane >> 4;
    const int b0 = blockIdx.x * BT;

    // init: build inp, zero pad columns of h1/h2 (cols >= 208 stay zero forever)
    for (int i = tid; i < BT * K1; i += THREADS) {
        int m = i / K1, c = i % K1;
        float v;
        if (c < DYN)            v = zdyn[(b0 + m) * DYN + c];
        else if (c < DYN + STA) v = zsta[(b0 + m) * STA + (c - DYN)];
        else if (c == DYN + STA) v = dtg[b0 + m];
        else                    v = 0.f;
        inp16[i] = (f16)v;
    }
    for (int i = tid; i < BT * K2; i += THREADS) { h1[i] = (f16)0.f; h2[i] = (f16)0.f; }
    __syncthreads();

    for (int t = 0; t < TSTEPS; ++t) {
        // P0: u*dt, clear accumulators
        for (int i = tid; i < BT * 32; i += THREADS) {
            int m = i >> 5, u = i & 31;
            float v = 0.f;
            if (u < U9) v = Ug[((long)t * BSZ + b0 + m) * U9 + u] * dtg[b0 + m];
            udt16[i] = (f16)v;
        }
        for (int i = tid; i < BT * DYN; i += THREADS) z2[i] = 0.f;
        for (int i = tid; i < BT * 16; i += THREADS) yacc[i] = 0.f;
        __syncthreads();

        // P1..P3: MLP
        mlp_gemm(inp16, K1, NK1, W1p, NT1, b1, HID, h1, K2, true, w, lane);
        __syncthreads();
        mlp_gemm(h1, K2, NK2, W2p, NT2, b2, HID, h2, K2, true, w, lane);
        __syncthreads();
        mlp_gemm(h2, K2, NK3, W3p, NT3, b3, HZD, hz16, HZD, false, w, lane);
        __syncthreads();

        // P4: z_next = (hz (x) z) @ At' + (hz (x) u*dt) @ Bt'
        // Each wave owns one K-quarter (kq = w&3) and 3 N-tiles; each B-fragment
        // load feeds BOTH M-tiles (register B-reuse -> half the L2 traffic).
        {
            const int kq = w & 3;
            v16h zf[2][3], hf[2][4], uf[2];
#pragma unroll
            for (int mt = 0; mt < 2; ++mt) {
                const int mrow = mt * 16 + lrow;
#pragma unroll
                for (int s = 0; s < 3; ++s) zf[mt][s] = load_a16(&inp16[mrow * K1 + 32 * s], lhi);
#pragma unroll
                for (int s = 0; s < 4; ++s) hf[mt][s] = load_a16(&hz16[mrow * HZD + 32 * s], lhi);
                uf[mt] = load_a16(&udt16[mrow * 32], lhi);
            }
            for (int nt = (w >> 2); nt < 6; nt += 2) {
                v8f acc0 = {}, acc1 = {};
                // At main: k in [kq*32, kq*32+32)
                for (int k = kq * 32; k < kq * 32 + 32; ++k) {
                    f16 hs0 = hz16[lrow * HZD + k];
                    f16 hs1 = hz16[(16 + lrow) * HZD + k];
#pragma unroll
                    for (int s = 0; s < 3; ++s) {
                        const f16* bp = Atp + ((long)nt * NKA + (k * 3 + s)) * 512 + lane * 16;
                        v16h b = *(const v16h*)bp;
                        acc0 = wmma32(scale16(zf[0][s], hs0), b, acc0);
                        acc1 = wmma32(scale16(zf[1][s], hs1), b, acc1);
                    }
                }
                if (kq == 3) {                    // At bias row (k == 128, hz == 1)
#pragma unroll
                    for (int s = 0; s < 3; ++s) {
                        const f16* bp = Atp + ((long)nt * NKA + (128 * 3 + s)) * 512 + lane * 16;
                        v16h b = *(const v16h*)bp;
                        acc0 = wmma32(zf[0][s], b, acc0);
                        acc1 = wmma32(zf[1][s], b, acc1);
                    }
                }
                // Bt main: slices [kq*9, kq*9+9)
                for (int sl = kq * 9; sl < kq * 9 + 9; ++sl) {
                    int u = sl >> 2, ks = sl & 3;
                    const f16* bp = Btp + ((long)nt * NKB + sl) * 512 + lane * 16;
                    v16h b = *(const v16h*)bp;
                    acc0 = wmma32(scale16(hf[0][ks], udt16[lrow * 32 + u]), b, acc0);
                    acc1 = wmma32(scale16(hf[1][ks], udt16[(16 + lrow) * 32 + u]), b, acc1);
                }
                if (kq == 3) {                    // Bt bias slice (36)
                    const f16* bp = Btp + ((long)nt * NKB + 36) * 512 + lane * 16;
                    v16h b = *(const v16h*)bp;
                    acc0 = wmma32(uf[0], b, acc0);
                    acc1 = wmma32(uf[1], b, acc1);
                }
                int n = nt * 16 + lrow;
#pragma unroll
                for (int v = 0; v < 8; ++v) {
                    atomicAdd(&z2[(v + 8 * lhi) * DYN + n], acc0[v]);       // ds_add_f32
                    atomicAdd(&z2[(16 + v + 8 * lhi) * DYN + n], acc1[v]);
                }
            }
        }
        __syncthreads();

        // P5: y = (hz (x) z_next) @ Ct' + (hz (x) u*dt) @ Dt'
        // 8-way K split (kq = w); each wave covers both M-tiles per B load.
        {
            const int kq = w;
            v16h zf[2][3], hf[2][4], uf[2];
#pragma unroll
            for (int mt = 0; mt < 2; ++mt) {
                const int mrow = mt * 16 + lrow;
#pragma unroll
                for (int s = 0; s < 3; ++s) zf[mt][s] = load_a32(&z2[mrow * DYN + 32 * s], lhi);
#pragma unroll
                for (int s = 0; s < 4; ++s) hf[mt][s] = load_a16(&hz16[mrow * HZD + 32 * s], lhi);
                uf[mt] = load_a16(&udt16[mrow * 32], lhi);
            }
            v8f acc0 = {}, acc1 = {};
            // Ct main: k in [kq*16, kq*16+16)
            for (int k = kq * 16; k < kq * 16 + 16; ++k) {
                f16 hs0 = hz16[lrow * HZD + k];
                f16 hs1 = hz16[(16 + lrow) * HZD + k];
#pragma unroll
                for (int s = 0; s < 3; ++s) {
                    const f16* bp = Ctp + ((long)(k * 3 + s)) * 512 + lane * 16;
                    v16h b = *(const v16h*)bp;
                    acc0 = wmma32(scale16(zf[0][s], hs0), b, acc0);
                    acc1 = wmma32(scale16(zf[1][s], hs1), b, acc1);
                }
            }
            if (kq == 7) {                        // Ct bias row (k == 128)
#pragma unroll
                for (int s = 0; s < 3; ++s) {
                    const f16* bp = Ctp + ((long)(128 * 3 + s)) * 512 + lane * 16;
                    v16h b = *(const v16h*)bp;
                    acc0 = wmma32(zf[0][s], b, acc0);
                    acc1 = wmma32(zf[1][s], b, acc1);
                }
            }
            // Dt main: slice range [kq*36/8, (kq+1)*36/8)
            int s0 = (kq * 36) >> 3, s1 = ((kq + 1) * 36) >> 3;
            for (int sl = s0; sl < s1; ++sl) {
                int u = sl >> 2, ks = sl & 3;
                const f16* bp = Dtp + ((long)sl) * 512 + lane * 16;
                v16h b = *(const v16h*)bp;
                acc0 = wmma32(scale16(hf[0][ks], udt16[lrow * 32 + u]), b, acc0);
                acc1 = wmma32(scale16(hf[1][ks], udt16[(16 + lrow) * 32 + u]), b, acc1);
            }
            if (kq == 7) {                        // Dt bias slice (36)
                const f16* bp = Dtp + ((long)36) * 512 + lane * 16;
                v16h b = *(const v16h*)bp;
                acc0 = wmma32(uf[0], b, acc0);
                acc1 = wmma32(uf[1], b, acc1);
            }
#pragma unroll
            for (int v = 0; v < 8; ++v) {
                atomicAdd(&yacc[(v + 8 * lhi) * 16 + lrow], acc0[v]);
                atomicAdd(&yacc[(16 + v + 8 * lhi) * 16 + lrow], acc1[v]);
            }
        }
        __syncthreads();

        // P6: write Z & Y, feed z_next back into inp for next step
        for (int i = tid; i < BT * DYN; i += THREADS) {
            int m = i / DYN, n = i % DYN;
            float v = z2[i];
            Zout[((long)t * BSZ + b0 + m) * DYN + n] = v;
            inp16[m * K1 + n] = (f16)v;
        }
        for (int i = tid; i < BT * NOBS; i += THREADS) {
            int m = i / NOBS, o = i % NOBS;
            Yout[((long)t * BSZ + b0 + m) * NOBS + o] = yacc[m * 16 + o];
        }
        __syncthreads();
    }
}

// ---------------- launcher ----------------

extern "C" void kernel_launch(void* const* d_in, const int* in_sizes, int n_in,
                              void* d_out, int out_size, void* d_ws, size_t ws_size,
                              hipStream_t stream) {
    const float* zdyn = (const float*)d_in[0];
    const float* zsta = (const float*)d_in[1];
    const float* dtg  = (const float*)d_in[2];
    const float* Ug   = (const float*)d_in[3];
    const float* W1   = (const float*)d_in[4];
    const float* b1   = (const float*)d_in[5];
    const float* W2   = (const float*)d_in[6];
    const float* b2   = (const float*)d_in[7];
    const float* W3   = (const float*)d_in[8];
    const float* b3   = (const float*)d_in[9];
    const float* AtW  = (const float*)d_in[10];
    const float* Atb  = (const float*)d_in[11];
    const float* BtW  = (const float*)d_in[12];
    const float* Btb  = (const float*)d_in[13];
    const float* CtW  = (const float*)d_in[14];
    const float* Ctb  = (const float*)d_in[15];
    const float* DtW  = (const float*)d_in[16];
    const float* Dtb  = (const float*)d_in[17];
    (void)in_sizes; (void)n_in; (void)out_size; (void)ws_size;

    f16* ws  = (f16*)d_ws;
    f16* W1p = ws;
    f16* W2p = W1p + SZ_W1;
    f16* W3p = W2p + SZ_W2;
    f16* Atp = W3p + SZ_W3;
    f16* Btp = Atp + SZ_AT;
    f16* Ctp = Btp + SZ_BT;
    f16* Dtp = Ctp + SZ_CT;

    pack_plain <<<SZ_W1 / 256, 256, 0, stream>>>(W1, W1p, NK1, 129, 200);
    pack_plain <<<SZ_W2 / 256, 256, 0, stream>>>(W2, W2p, NK2, 200, 200);
    pack_plain <<<SZ_W3 / 256, 256, 0, stream>>>(W3, W3p, NK3, 200, 128);
    pack_bilin_j<<<SZ_AT / 256, 256, 0, stream>>>(AtW, Atb, Atp, NKA, 128, 96);
    pack_bilin_u<<<SZ_BT / 256, 256, 0, stream>>>(BtW, Btb, Btp, NKB, 96);
    pack_bilin_j<<<SZ_CT / 256, 256, 0, stream>>>(CtW, Ctb, Ctp, NKC, 128, 14);
    pack_bilin_u<<<SZ_DT / 256, 256, 0, stream>>>(DtW, Dtb, Dtp, NKD, 14);

    float* Zout = (float*)d_out;
    float* Yout = Zout + (long)TSTEPS * BSZ * DYN;
    cond_lin_main<<<BSZ / BT, THREADS, 0, stream>>>(zdyn, zsta, dtg, Ug, b1, b2, b3,
                                                    W1p, W2p, W3p, Atp, Btp, Ctp, Dtp,
                                                    Zout, Yout);
}
